// MeshAutoencoder_56968446214232
// MI455X (gfx1250) — matrix operand order repaired and checked by reading, not compiled, not run on hardware
//
#include <hip/hip_runtime.h>
#include <hip/hip_bf16.h>
#include <stdint.h>

// ---- problem constants -----------------------------------------------------
#define B_      8
#define NV_     2048
#define NF_     4096
#define E_      12288
#define DIM_    512
#define DCE_    64
#define DCB_    192
#define KQ_     128
#define NQ_     2
#define BITS_   14

static constexpr int  M_FACE = B_ * NF_;   // 32768
static constexpr int  M_PAD  = M_FACE + 1; // 32769 (graph pad node)
static constexpr int  NVP    = NV_ + 1;    // 2049  (vertex pad segment)

typedef __attribute__((ext_vector_type(16))) __bf16 v16bf;
typedef __attribute__((ext_vector_type(8)))  float  v8f;

__device__ __forceinline__ unsigned short f2bf(float f) {
  union { float f; unsigned int u; } x; x.f = f;
  unsigned int u = x.u;
  unsigned int r = (u + 0x7FFFu + ((u >> 16) & 1u)) >> 16; // RNE truncate
  return (unsigned short)r;
}

// one per-lane 16B async copy global -> LDS (ASYNCcnt-tracked)
__device__ __forceinline__ void async_b128(unsigned lds_addr, const void* gaddr) {
  asm volatile("global_load_async_to_lds_b128 %0, %1, off"
               :: "v"(lds_addr), "v"(gaddr) : "memory");
}
// drain all async copies
__device__ __forceinline__ void wait_async_0() {
  asm volatile("s_wait_asynccnt 0x0" ::: "memory");
}
// allow the 12 just-issued copies of the NEXT tile to stay in flight; the
// older group (this tile's 12) must have completed (ASYNCcnt is per-wave,
// completes in order).
__device__ __forceinline__ void wait_async_12() {
  asm volatile("s_wait_asynccnt 0xc" ::: "memory");
}

// stage one 128x64 A tile + 64x64 B tile (bf16) into LDS via async DMA.
// OOB A rows are clamped to M-1 (their outputs are masked at the store) so
// the hot loop stays free of EXEC-mask branching.
__device__ __forceinline__ void stage_tiles(const unsigned short* __restrict__ A,
                                            const unsigned short* __restrict__ Bt,
                                            unsigned short* As, unsigned short* Bs,
                                            int rowBase, int colBase, int kk,
                                            int M, int K, int tid) {
  #pragma unroll
  for (int q = 0; q < 8; ++q) {            // A: 1024 chunks of 16B, 8/thread
    int c   = tid + q * 128;
    int row = c >> 3;
    int sub = (c & 7) * 8;                 // ushort offset in row (stride 64)
    int grow = rowBase + row; if (grow >= M) grow = M - 1;
    async_b128((unsigned)(uintptr_t)(As + row * 64 + sub),
               A + (size_t)grow * K + kk + sub);
    if (q == 0 && kk + 64 < K)             // speculative prefetch of tile+2
      __builtin_prefetch(A + (size_t)grow * K + kk + 64 + sub, 0, 0);
  }
  #pragma unroll
  for (int q = 0; q < 4; ++q) {            // B: 512 chunks of 16B, 4/thread
    int c   = tid + q * 128;
    int row = c >> 3;
    int sub = (c & 7) * 8;
    async_b128((unsigned)(uintptr_t)(Bs + row * 64 + sub),
               Bt + (size_t)(colBase + row) * K + kk + sub);
  }
}

// ---- generic bf16 WMMA GEMM ------------------------------------------------
// C[M,N](f32) = A[M,K](bf16 row-major) * Bt[N,K](bf16, B transposed)
// flags: bit0 = add bias[N], bit1 = accumulate into existing C
// Block: 128 threads = 4 waves; tile 128x64, K-step 64, double-buffered LDS.
// Wave w owns rows [32w,32w+32) x all 64 cols: 2 x (2x4) v_wmma per K-step.
// Async copies of tile t+1 overlap the 16 WMMAs of tile t.
__launch_bounds__(128)
__global__ void k_gemm(const unsigned short* __restrict__ A,
                       const unsigned short* __restrict__ Bt,
                       const float* __restrict__ bias,
                       float* __restrict__ C,
                       int M, int N, int K, int flags)
{
  __shared__ __align__(16) unsigned short As[2][128 * 64]; // 2x16KB [m][k]
  __shared__ __align__(16) unsigned short Bs[2][64 * 64];  // 2x8KB  [n][k]

  const int tid  = threadIdx.x;
  const int lane = tid & 31;
  const int wave = tid >> 5;
  const int rowBase = blockIdx.y * 128;
  const int colBase = blockIdx.x * 64;
  const int wm = wave * 32;

  __builtin_amdgcn_s_wait_tensorcnt(0); // gfx1250 split-counter path (no-op here)

  const v8f vzero = {0.f, 0.f, 0.f, 0.f, 0.f, 0.f, 0.f, 0.f};
  v8f acc[2][4];
  #pragma unroll
  for (int i = 0; i < 2; ++i)
    #pragma unroll
    for (int j = 0; j < 4; ++j) acc[i][j] = vzero;

  union Frag { uint4 u[2]; v16bf v; };

  // prologue: stage tile 0 into buffer 0
  stage_tiles(A, Bt, As[0], Bs[0], rowBase, colBase, 0, M, K, tid);

  for (int kk = 0; kk < K; kk += 64) {
    const int cur = (kk >> 6) & 1;
    if (kk + 64 < K) {
      // overlap: launch next tile's copies into the other buffer, then wait
      // only for THIS tile's group (older 12) to have landed.
      stage_tiles(A, Bt, As[cur ^ 1], Bs[cur ^ 1], rowBase, colBase, kk + 64, M, K, tid);
      wait_async_12();
    } else {
      wait_async_0();
    }
    __syncthreads();

    const unsigned short* Ac = As[cur];
    const unsigned short* Bc = Bs[cur];
    #pragma unroll
    for (int ks = 0; ks < 64; ks += 32) {
      // A fragment (16x32): lanes 0-15 M=lane,kb=0; 16-31 M=lane-16,kb=8.
      // slots 0-7 = K kb..kb+7, slots 8-15 = K kb+16..kb+23 (ISA 7.12.2)
      Frag a[2], b[4];
      int arow = lane & 15;
      int kb   = (lane >> 4) * 8;
      #pragma unroll
      for (int i = 0; i < 2; ++i) {
        a[i].u[0] = *(const uint4*)(Ac + (wm + 16 * i + arow) * 64 + ks + kb);
        a[i].u[1] = *(const uint4*)(Ac + (wm + 16 * i + arow) * 64 + ks + kb + 16);
      }
      // B fragment (32x16): lanes 0-15 N=lane,kb=0; 16-31 N=lane-16,kb=8; slots 0-7 = K kb..kb+7, 8-15 = K kb+16..kb+23
      int bcol = lane & 15;
      int kbb  = (lane >> 4) * 8;
      #pragma unroll
      for (int j = 0; j < 4; ++j) {
        b[j].u[0] = *(const uint4*)(Bc + (16 * j + bcol) * 64 + ks + kbb);
        b[j].u[1] = *(const uint4*)(Bc + (16 * j + bcol) * 64 + ks + kbb + 16);
      }
      #pragma unroll
      for (int i = 0; i < 2; ++i)
        #pragma unroll
        for (int j = 0; j < 4; ++j)
          acc[i][j] = __builtin_amdgcn_wmma_f32_16x16x32_bf16(
              false, a[i].v, false, b[j].v, (short)0, acc[i][j], false, false);
    }
    __syncthreads(); // all waves done reading buf 'cur' before it is restaged
  }

  // C/D layout: VGPR r -> M = r + 8*(lane>=16), N = lane&15 (ISA 7.12.2)
  #pragma unroll
  for (int i = 0; i < 2; ++i) {
    #pragma unroll
    for (int j = 0; j < 4; ++j) {
      int ccol  = colBase + 16 * j + (lane & 15);
      int rbase = rowBase + wm + 16 * i + 8 * (lane >> 4);
      float bb = ((flags & 1) && ccol < N) ? bias[ccol] : 0.f;
      #pragma unroll
      for (int r = 0; r < 8; ++r) {
        int row = rbase + r;
        if (row < M && ccol < N) {
          float v = acc[i][j][r] + bb;
          if (flags & 2) v += C[(size_t)row * N + ccol];
          C[(size_t)row * N + ccol] = v;
        }
      }
    }
  }
}

// ---- weight / activation conversion ---------------------------------------
__global__ void k_f32_to_bf16(const float* __restrict__ s, unsigned short* __restrict__ d, long n) {
  long i = (long)blockIdx.x * blockDim.x + threadIdx.x;
  if (i < n) d[i] = f2bf(s[i]);
}

// W[K][N] f32 -> Bt[N][K] bf16
__global__ void k_w_transpose(const float* __restrict__ W, unsigned short* __restrict__ Bt, int K, int N) {
  long i = (long)blockIdx.x * blockDim.x + threadIdx.x;
  if (i >= (long)N * K) return;
  int n = (int)(i / K), k = (int)(i - (long)n * K);
  Bt[i] = f2bf(W[(size_t)k * N + n]);
}

// conv weight w[O=512][I=512][T=3] -> Bt[o][t*512+i]
__global__ void k_conv_w(const float* __restrict__ w, unsigned short* __restrict__ Bt) {
  long i = (long)blockIdx.x * blockDim.x + threadIdx.x;
  if (i >= (long)512 * 1536) return;
  int o = (int)(i / 1536), col = (int)(i - (long)o * 1536);
  int t = col / 512, ch = col - t * 512;
  Bt[i] = f2bf(w[((size_t)o * 512 + ch) * 3 + t]);
}

// ---- front end: discretize + embedding gather ------------------------------
__global__ void k_discretize(const float* __restrict__ v, int* __restrict__ disc, long n) {
  long i = (long)blockIdx.x * blockDim.x + threadIdx.x;
  if (i >= n) return;
  float t = (v[i] + 1.f) * 0.5f * (float)KQ_ - 0.5f;
  float r = fminf(fmaxf(rintf(t), 0.f), (float)(KQ_ - 1));
  disc[i] = (int)r;
}

__global__ void k_embed(const int* __restrict__ disc, const int* __restrict__ faces,
                        const float* __restrict__ ce, unsigned short* __restrict__ A) {
  long i = (long)blockIdx.x * blockDim.x + threadIdx.x;
  if (i >= (long)M_FACE * 576) return;
  int row = (int)(i / 576), col = (int)(i - (long)row * 576);
  int b = row >> 12, nf = row & 4095;
  int j = col >> 6, c = col & 63;        // col = j*64 + c, j = corner*3 + axis
  int f    = faces[((size_t)(b * NF_ + nf)) * 3 + (j / 3)];
  int code = disc[((size_t)(b * NV_ + f)) * 3 + (j % 3)];
  A[i] = f2bf(ce[code * DCE_ + c]);
}

// ---- SAGE aggregation ------------------------------------------------------
__launch_bounds__(128)
__global__ void k_edge_agg(const float* __restrict__ x, const int* __restrict__ fedge,
                           const unsigned char* __restrict__ emask,
                           float* __restrict__ agg, float* __restrict__ cnt) {
  int e = blockIdx.x;            // 0 .. B*E-1
  int b = e / E_, t = e - b * E_;
  int s, d;
  if (emask[(size_t)b * E_ + t]) {
    s = fedge[((size_t)b * 2 + 0) * E_ + t] + b * NF_;
    d = fedge[((size_t)b * 2 + 1) * E_ + t] + b * NF_;
  } else { s = M_FACE; d = M_FACE; }
  const float* xs = x + (size_t)s * DIM_;
  float* ad = agg + (size_t)d * DIM_;
  for (int c = threadIdx.x; c < DIM_; c += 128) atomicAdd(&ad[c], xs[c]);
  if (threadIdx.x == 0) atomicAdd(&cnt[d], 1.f);
}

__global__ void k_divcnt(float* __restrict__ agg, const float* __restrict__ cnt, long n) {
  long i = (long)blockIdx.x * blockDim.x + threadIdx.x;
  if (i < n) agg[i] /= fmaxf(cnt[i >> 9], 1.f);
}

// ---- vertex scatter-mean + LFQ ---------------------------------------------
__global__ void k_scatter_den(const int* __restrict__ faces, const unsigned char* __restrict__ fmask,
                              float* __restrict__ den) {
  long i = (long)blockIdx.x * blockDim.x + threadIdx.x;
  if (i >= (long)B_ * NF_ * 3) return;
  int b = (int)(i / (NF_ * 3)), f3 = (int)(i - (long)b * NF_ * 3);
  int nf = f3 / 3, corner = f3 - nf * 3;
  int v = fmask[(size_t)b * NF_ + nf] ? faces[((size_t)(b * NF_ + nf)) * 3 + corner] : NV_;
  atomicAdd(&den[(size_t)b * NVP + v], 1.f);
}

__global__ void k_scatter_num(const float* __restrict__ fe, const int* __restrict__ faces,
                              const unsigned char* __restrict__ fmask, float* __restrict__ num) {
  long i = (long)blockIdx.x * blockDim.x + threadIdx.x;
  if (i >= (long)B_ * NF_ * 3 * DCB_) return;
  long row = i / DCB_;
  int c = (int)(i - row * DCB_);
  int b = (int)(row / (NF_ * 3)), f3 = (int)(row - (long)b * NF_ * 3);
  int nf = f3 / 3, corner = f3 - nf * 3;
  int v = fmask[(size_t)b * NF_ + nf] ? faces[((size_t)(b * NF_ + nf)) * 3 + corner] : NV_;
  atomicAdd(&num[((size_t)(b * NVP + v)) * DCB_ + c],
            fe[((size_t)(b * NF_ + nf)) * 576 + corner * DCB_ + c]);
}

__global__ void k_avg(float* __restrict__ num, const float* __restrict__ den, long n) {
  long i = (long)blockIdx.x * blockDim.x + threadIdx.x;
  if (i < n) num[i] /= fmaxf(den[i / DCB_], 1e-5f);
}

__launch_bounds__(192)
__global__ void k_lfq(const float* __restrict__ avg, float* __restrict__ quant,
                      const float* __restrict__ lin, const float* __restrict__ lout) {
  __shared__ float sres[DCB_];
  __shared__ float sbit[16];
  int row = blockIdx.x;          // 0 .. B*NVP-1
  int c = threadIdx.x;
  sres[c] = avg[(size_t)row * DCB_ + c];
  float qacc = 0.f;
  __syncthreads();
  for (int qi = 0; qi < NQ_; ++qi) {
    if (c < BITS_) {
      float h = 0.f;
      for (int k = 0; k < DCB_; ++k) h += sres[k] * lin[((size_t)qi * DCB_ + k) * BITS_ + c];
      sbit[c] = (h > 0.f) ? 1.f : -1.f;
    }
    __syncthreads();
    float o = 0.f;
    for (int j = 0; j < BITS_; ++j) o += sbit[j] * lout[((size_t)qi * BITS_ + j) * DCB_ + c];
    qacc += o;
    sres[c] -= o;
    __syncthreads();
  }
  quant[(size_t)row * DCB_ + c] = qacc;
}

__global__ void k_gather_feo(const float* __restrict__ quant, const int* __restrict__ faces,
                             const unsigned char* __restrict__ fmask, unsigned short* __restrict__ A) {
  long i = (long)blockIdx.x * blockDim.x + threadIdx.x;
  if (i >= (long)M_FACE * 576) return;
  int row = (int)(i / 576), col = (int)(i - (long)row * 576);
  int b = row >> 12, nf = row & 4095;
  int corner = col / DCB_, c = col - corner * DCB_;
  int v = fmask[(size_t)b * NF_ + nf] ? faces[((size_t)(b * NF_ + nf)) * 3 + corner] : NV_;
  A[i] = f2bf(quant[((size_t)(b * NVP + v)) * DCB_ + c]);
}

// ---- decoder: im2col, groupnorm+silu, residual -----------------------------
__global__ void k_im2col(const float* __restrict__ x, const unsigned char* __restrict__ fmask,
                         unsigned short* __restrict__ A) {
  long i = (long)blockIdx.x * blockDim.x + threadIdx.x;
  if (i >= (long)M_FACE * 1536) return;
  int row = (int)(i / 1536), col = (int)(i - (long)row * 1536);
  int t = col / 512, ch = col - t * 512;
  int b = row >> 12, n = row & 4095;
  int n2 = n + t - 1;
  float v = 0.f;
  if (n2 >= 0 && n2 < NF_) {
    float m = fmask[(size_t)b * NF_ + n2] ? 1.f : 0.f;   // x = x*m before conv
    v = x[((size_t)(b * NF_ + n2)) * 512 + ch] * m;
  }
  A[i] = f2bf(v);
}

__launch_bounds__(256)
__global__ void k_gn_silu(float* __restrict__ y, const unsigned char* __restrict__ fmask,
                          const float* __restrict__ gamma, const float* __restrict__ beta) {
  int b = blockIdx.x >> 3;       // / GROUPS
  int g = blockIdx.x & 7;
  const int CG = 64;
  __shared__ float ssum[256], ssq[256];
  float s = 0.f, q = 0.f;
  for (int idx = threadIdx.x; idx < CG * NF_; idx += 256) {
    int n = idx >> 6, c = g * 64 + (idx & 63);
    float m = fmask[(size_t)b * NF_ + n] ? 1.f : 0.f;    // x = conv_out * m
    float v = y[((size_t)(b * NF_ + n)) * 512 + c] * m;
    s += v; q += v * v;
  }
  ssum[threadIdx.x] = s; ssq[threadIdx.x] = q;
  __syncthreads();
  for (int off = 128; off > 0; off >>= 1) {
    if (threadIdx.x < off) { ssum[threadIdx.x] += ssum[threadIdx.x + off]; ssq[threadIdx.x] += ssq[threadIdx.x + off]; }
    __syncthreads();
  }
  float inv = 1.f / (float)(CG * NF_);
  float mean = ssum[0] * inv;
  float var  = ssq[0] * inv - mean * mean;
  float rstd = rsqrtf(var + 1e-5f);
  for (int idx = threadIdx.x; idx < CG * NF_; idx += 256) {
    int n = idx >> 6, c = g * 64 + (idx & 63);
    size_t o = ((size_t)(b * NF_ + n)) * 512 + c;
    float m = fmask[(size_t)b * NF_ + n] ? 1.f : 0.f;
    float v = (y[o] * m - mean) * rstd * gamma[c] + beta[c];
    y[o] = v * (1.f / (1.f + __expf(-v)));               // SiLU
  }
}

__global__ void k_resadd(float* __restrict__ xd, const float* __restrict__ h, long n) {
  long i = (long)blockIdx.x * blockDim.x + threadIdx.x;
  if (i < n) xd[i] += h[i];
}

// ---- host-side orchestration ----------------------------------------------
static inline int cdiv(long a, long b) { return (int)((a + b - 1) / b); }

extern "C" void kernel_launch(void* const* d_in, const int* in_sizes, int n_in,
                              void* d_out, int out_size, void* d_ws, size_t ws_size,
                              hipStream_t stream) {
  (void)in_sizes; (void)n_in; (void)out_size; (void)ws_size;

  const float*         vertices  = (const float*)d_in[0];
  const int*           faces     = (const int*)d_in[1];
  const int*           fedges    = (const int*)d_in[2];
  const unsigned char* emask     = (const unsigned char*)d_in[3];
  const unsigned char* fmask     = (const unsigned char*)d_in[4];
  const float*         coor_emb  = (const float*)d_in[5];
  const float*         W_in      = (const float*)d_in[6];
  const float*         b_in      = (const float*)d_in[7];
  const float*         sage_Wl   = (const float*)d_in[8];
  const float*         sage_Wr   = (const float*)d_in[9];
  const float*         sage_b    = (const float*)d_in[10];
  const float*         W_cb      = (const float*)d_in[11];
  const float*         b_cb      = (const float*)d_in[12];
  const float*         lfq_in    = (const float*)d_in[13];
  const float*         lfq_out   = (const float*)d_in[14];
  const float*         W_out     = (const float*)d_in[15];
  const float*         b_out     = (const float*)d_in[16];
  const float*         conv_w    = (const float*)d_in[17];
  const float*         conv_b    = (const float*)d_in[18];
  const float*         gn_g      = (const float*)d_in[19];
  const float*         gn_b      = (const float*)d_in[20];
  const float*         W_logits  = (const float*)d_in[21];
  const float*         b_logits  = (const float*)d_in[22];
  float*               out       = (float*)d_out;

  // workspace carve (256B aligned)
  char* ws = (char*)d_ws;
  size_t off = 0;
  auto take = [&](size_t bytes) -> void* {
    void* p = ws + off; off += (bytes + 255) & ~(size_t)255; return p;
  };
  unsigned short* bfA   = (unsigned short*)take((size_t)M_FACE * 1536 * 2); // 96MB, multi-use
  float* xf   = (float*)take((size_t)M_PAD * 512 * 4);
  float* yf   = (float*)take((size_t)M_PAD * 512 * 4);
  float* cnt  = (float*)take((size_t)M_PAD * 4);
  float* fe   = (float*)take((size_t)M_FACE * 576 * 4);
  float* avg  = (float*)take((size_t)B_ * NVP * DCB_ * 4);
  float* den  = (float*)take((size_t)B_ * NVP * 4);
  float* quant= (float*)take((size_t)B_ * NVP * DCB_ * 4);
  int*   disc = (int*)take((size_t)B_ * NV_ * 3 * 4);
  unsigned short* wt_in  = (unsigned short*)take((size_t)512 * 576 * 2);
  unsigned short* wt_l   = (unsigned short*)take((size_t)2 * 512 * 512 * 2);
  unsigned short* wt_r   = (unsigned short*)take((size_t)2 * 512 * 512 * 2);
  unsigned short* wt_cb  = (unsigned short*)take((size_t)576 * 512 * 2);
  unsigned short* wt_out = (unsigned short*)take((size_t)512 * 576 * 2);
  unsigned short* wt_cv  = (unsigned short*)take((size_t)4 * 512 * 1536 * 2);
  unsigned short* wt_log = (unsigned short*)take((size_t)1152 * 512 * 2);
  // SAGE-phase sub-regions inside bfA (disjoint)
  unsigned short* xbf   = bfA;
  unsigned short* aggbf = bfA + (size_t)25165824; // +48MB

  const int T = 256;

  // weights -> bf16 [N][K]
  k_w_transpose<<<cdiv((long)512 * 576, T), T, 0, stream>>>(W_in, wt_in, 576, 512);
  k_w_transpose<<<cdiv((long)512 * 512, T), T, 0, stream>>>(sage_Wl,               wt_l,                512, 512);
  k_w_transpose<<<cdiv((long)512 * 512, T), T, 0, stream>>>(sage_Wl + 512 * 512,   wt_l + 512 * 512,    512, 512);
  k_w_transpose<<<cdiv((long)512 * 512, T), T, 0, stream>>>(sage_Wr,               wt_r,                512, 512);
  k_w_transpose<<<cdiv((long)512 * 512, T), T, 0, stream>>>(sage_Wr + 512 * 512,   wt_r + 512 * 512,    512, 512);
  k_w_transpose<<<cdiv((long)576 * 512, T), T, 0, stream>>>(W_cb, wt_cb, 512, 576);
  k_w_transpose<<<cdiv((long)512 * 576, T), T, 0, stream>>>(W_out, wt_out, 576, 512);
  k_w_transpose<<<cdiv((long)1152 * 512, T), T, 0, stream>>>(W_logits, wt_log, 512, 1152);
  for (int s = 0; s < 4; ++s)
    k_conv_w<<<cdiv((long)512 * 1536, T), T, 0, stream>>>(conv_w + (size_t)s * 512 * 512 * 3,
                                                          wt_cv + (size_t)s * 512 * 1536);

  // Phase 1: embed + W_in
  k_discretize<<<cdiv((long)B_ * NV_ * 3, T), T, 0, stream>>>(vertices, disc, (long)B_ * NV_ * 3);
  k_embed<<<cdiv((long)M_FACE * 576, T), T, 0, stream>>>(disc, faces, coor_emb, bfA);
  k_gemm<<<dim3(512 / 64, cdiv(M_FACE, 128)), 128, 0, stream>>>(bfA, wt_in, b_in, xf, M_FACE, 512, 576, 1);
  hipMemsetAsync(xf + (size_t)M_FACE * 512, 0, 512 * sizeof(float), stream); // pad node

  // Phase 2: 2x SAGEConv (mean agg)
  for (int l = 0; l < 2; ++l) {
    k_f32_to_bf16<<<cdiv((long)M_PAD * 512, T), T, 0, stream>>>(xf, xbf, (long)M_PAD * 512);
    hipMemsetAsync(yf, 0, (size_t)M_PAD * 512 * sizeof(float), stream);
    hipMemsetAsync(cnt, 0, (size_t)M_PAD * sizeof(float), stream);
    k_edge_agg<<<B_ * E_, 128, 0, stream>>>(xf, fedges, emask, yf, cnt);
    k_divcnt<<<cdiv((long)M_PAD * 512, T), T, 0, stream>>>(yf, cnt, (long)M_PAD * 512);
    k_f32_to_bf16<<<cdiv((long)M_PAD * 512, T), T, 0, stream>>>(yf, aggbf, (long)M_PAD * 512);
    k_gemm<<<dim3(8, cdiv(M_PAD, 128)), 128, 0, stream>>>(aggbf, wt_l + (size_t)l * 512 * 512, nullptr, xf, M_PAD, 512, 512, 0);
    k_gemm<<<dim3(8, cdiv(M_PAD, 128)), 128, 0, stream>>>(xbf,   wt_r + (size_t)l * 512 * 512, sage_b + (size_t)l * 512, xf, M_PAD, 512, 512, 1 | 2);
  }

  // Phase 3: fe = x @ W_cb + b_cb
  k_f32_to_bf16<<<cdiv((long)M_FACE * 512, T), T, 0, stream>>>(xf, bfA, (long)M_FACE * 512);
  k_gemm<<<dim3(576 / 64, cdiv(M_FACE, 128)), 128, 0, stream>>>(bfA, wt_cb, b_cb, fe, M_FACE, 576, 512, 1);

  // Phase 4: scatter-mean onto vertices, LFQ, gather back, W_out
  hipMemsetAsync(avg, 0, (size_t)B_ * NVP * DCB_ * sizeof(float), stream);
  hipMemsetAsync(den, 0, (size_t)B_ * NVP * sizeof(float), stream);
  k_scatter_den<<<cdiv((long)B_ * NF_ * 3, T), T, 0, stream>>>(faces, fmask, den);
  k_scatter_num<<<cdiv((long)B_ * NF_ * 3 * DCB_, T), T, 0, stream>>>(fe, faces, fmask, avg);
  k_avg<<<cdiv((long)B_ * NVP * DCB_, T), T, 0, stream>>>(avg, den, (long)B_ * NVP * DCB_);
  k_lfq<<<B_ * NVP, DCB_, 0, stream>>>(avg, quant, lfq_in, lfq_out);
  k_gather_feo<<<cdiv((long)M_FACE * 576, T), T, 0, stream>>>(quant, faces, fmask, bfA);
  k_gemm<<<dim3(8, cdiv(M_FACE, 128)), 128, 0, stream>>>(bfA, wt_out, b_out, xf, M_FACE, 512, 576, 1);

  // Phase 5: conv-resnet decoder (2 blocks x 2 sub-blocks)
  for (int blk = 0; blk < 2; ++blk) {
    for (int sub = 0; sub < 2; ++sub) {
      const float* cur = (sub == 0) ? xf : yf;
      int wi = blk * 2 + sub;
      k_im2col<<<cdiv((long)M_FACE * 1536, T), T, 0, stream>>>(cur, fmask, bfA);
      k_gemm<<<dim3(8, cdiv(M_FACE, 128)), 128, 0, stream>>>(bfA, wt_cv + (size_t)wi * 512 * 1536,
                                                             conv_b + (size_t)wi * 512, yf, M_FACE, 512, 1536, 1);
      k_gn_silu<<<B_ * 8, 256, 0, stream>>>(yf, fmask, gn_g + (size_t)wi * 512, gn_b + (size_t)wi * 512);
    }
    k_resadd<<<cdiv((long)M_FACE * 512, T), T, 0, stream>>>(xf, yf, (long)M_FACE * 512);
  }

  // Phase 6: logits
  k_f32_to_bf16<<<cdiv((long)M_FACE * 512, T), T, 0, stream>>>(xf, bfA, (long)M_FACE * 512);
  k_gemm<<<dim3(1152 / 64, cdiv(M_FACE, 128)), 128, 0, stream>>>(bfA, wt_log, b_logits, out, M_FACE, 1152, 512, 1);
}
